// VQVAE_55920474194575
// MI455X (gfx1250) — compile-verified, matrix operand order checked by reading, not run on hardware
//
#include <hip/hip_runtime.h>
#include <hip/hip_bf16.h>
#include <stdint.h>
#include <stddef.h>

// ---------------------------------------------------------------------------
// VQ-VAE forward for MI455X (gfx1250): bf16 WMMA GEMMs, f32 accumulate,
// async Global->LDS (ASYNCcnt) double-buffered pipeline, 256x128 block tiles.
// ---------------------------------------------------------------------------

#define BB    16384
#define DIN   1024
#define HH    2048
#define EE    256
#define KC    8192
#define EPSF  1e-5f

typedef __bf16 bf16_t;
typedef __attribute__((ext_vector_type(16))) __bf16 v16bf;
typedef __attribute__((ext_vector_type(8)))  __bf16 v8bf;
typedef __attribute__((ext_vector_type(8)))  float  v8f;

__device__ __forceinline__ bf16_t f2bf(float f) {
    unsigned int u = __float_as_uint(f);
    unsigned int r = (u + 0x7FFFu + ((u >> 16) & 1u)) >> 16;
    unsigned short s = (unsigned short)r;
    return __builtin_bit_cast(bf16_t, s);
}

__device__ __forceinline__ unsigned int fkey(float f) {
    unsigned int u = __float_as_uint(f);
    return (u & 0x80000000u) ? ~u : (u | 0x80000000u);
}

// --- CDNA5 async Global->LDS copy (16B per lane), tracked by ASYNCcnt ------
__device__ __forceinline__ void cp_async16(void* lds_ptr, const void* gptr) {
    uint32_t lds_off = (uint32_t)(uintptr_t)lds_ptr;
    unsigned long long ga = (unsigned long long)(uintptr_t)gptr;
    asm volatile("global_load_async_to_lds_b128 %0, %1, off"
                 :: "v"(lds_off), "v"(ga) : "memory");
}

template<int N>
__device__ __forceinline__ void wait_async() {
    asm volatile("s_wait_asynccnt %0" :: "n"(N) : "memory");
}

// -------------------------------- converters -------------------------------

__global__ void k_f32_to_bf16(const float* __restrict__ in,
                              bf16_t* __restrict__ out, int n) {
    int i = (blockIdx.x * blockDim.x + threadIdx.x) * 4;
    if (i + 3 < n) {
        float4 v = *(const float4*)(in + i);
        out[i + 0] = f2bf(v.x);
        out[i + 1] = f2bf(v.y);
        out[i + 2] = f2bf(v.z);
        out[i + 3] = f2bf(v.w);
    }
}

// w: [K][N] f32 row-major  ->  wt: [N][K] bf16 row-major (K-major operand)
__global__ void k_transpose_bf16(const float* __restrict__ w,
                                 bf16_t* __restrict__ wt, int K, int N) {
    __shared__ float tile[32][33];
    const int kb = blockIdx.y * 32, nb = blockIdx.x * 32;
    const int tx = threadIdx.x, ty = threadIdx.y;   // 32 x 8
    for (int r = 0; r < 32; r += 8)
        tile[ty + r][tx] = w[(size_t)(kb + ty + r) * N + (nb + tx)];
    __syncthreads();
    for (int r = 0; r < 32; r += 8)
        wt[(size_t)(nb + ty + r) * K + (kb + tx)] = f2bf(tile[tx][ty + r]);
}

// codebook prep: bf16 copy + per-code squared norm
__global__ void k_cb_prep(const float* __restrict__ cb,
                          bf16_t* __restrict__ cbb,
                          float* __restrict__ cnorm) {
    const int k = blockIdx.x, t = threadIdx.x;      // 256 threads = E
    float v = cb[(size_t)k * EE + t];
    cbb[(size_t)k * EE + t] = f2bf(v);
    float sq = v * v;
    __shared__ float red[8];
    for (int s = 16; s > 0; s >>= 1) sq += __shfl_down(sq, s, 32);
    if ((t & 31) == 0) red[t >> 5] = sq;
    __syncthreads();
    if (t < 8) {
        float x = red[t];
        for (int s = 4; s > 0; s >>= 1) x += __shfl_down(x, s, 32);
        if (t == 0) cnorm[k] = x;
    }
}

__global__ void k_init_best(unsigned long long* __restrict__ best, int n) {
    int i = blockIdx.x * blockDim.x + threadIdx.x;
    if (i < n) best[i] = 0xFFFFFFFFFFFFFFFFull;
}

// ------------------------------ WMMA GEMM ----------------------------------
// C[M,N] = A[M,K] * B^T  with A, B both K-major bf16 (B stored [N,K]).
// 256 thr (8 waves), block tile 256x128, TK=32, double-buffered async LDS.
// Wave grid 4(M) x 2(N); wave tile 64x64 = 4x4 v_wmma_f32_16x16x32_bf16
// (16 WMMA per wave between barriers).
// EPI=0: out_bf16 = relu(gamma*(acc+bias-mean)*rsqrt(var+eps)+beta)
// EPI=1: out_f32 = acc + bias  (optional bf16 mirror if outb != nullptr)

#define TKK  32
#define LDT  40   // TK + 8 bf16 pad (80 B rows; 16B chunks stay aligned)
#define TM   256
#define TN   128

template<int EPI>
__global__ __launch_bounds__(256)
void k_gemm(const bf16_t* __restrict__ A, const bf16_t* __restrict__ Bm,
            int M, int N, int K,
            const float* __restrict__ bias,
            const float* __restrict__ gamma, const float* __restrict__ beta,
            const float* __restrict__ mean,  const float* __restrict__ var,
            float* __restrict__ outf, bf16_t* __restrict__ outb)
{
    __shared__ __align__(16) bf16_t As[2][TM * LDT];   // 2*256*40*2 = 40 KB
    __shared__ __align__(16) bf16_t Bs[2][TN * LDT];   // 2*128*40*2 = 20 KB

    const int mBase = blockIdx.y * TM;
    const int nBase = blockIdx.x * TN;
    const int tid   = threadIdx.x;
    const int lane  = tid & 31;
    const int wave  = tid >> 5;
    const int g     = lane >> 4;     // half-wave id (K-group select)
    const int lm    = lane & 15;     // row/col within fragment
    const int waveM = wave >> 1;     // 0..3  (64-row slice)
    const int waveN = wave & 1;      // 0..1  (64-col slice)

    v8f acc[4][4];
    for (int i = 0; i < 4; ++i)
        for (int j = 0; j < 4; ++j)
            acc[i][j] = (v8f){0.f,0.f,0.f,0.f,0.f,0.f,0.f,0.f};

    auto stage = [&](int kt, int buf) {            // 6 async ops / wave
        const int k0 = kt * TKK;
        #pragma unroll
        for (int i = 0; i < 4; ++i) {              // A: 1024 16B vecs
            int lin = i * 256 + tid;
            int row = lin >> 2, vec = (lin & 3) * 8;
            cp_async16(&As[buf][row * LDT + vec],
                       &A[(size_t)(mBase + row) * K + k0 + vec]);
        }
        #pragma unroll
        for (int i = 0; i < 2; ++i) {              // B: 512 16B vecs
            int lin = i * 256 + tid;
            int row = lin >> 2, vec = (lin & 3) * 8;
            cp_async16(&Bs[buf][row * LDT + vec],
                       &Bm[(size_t)(nBase + row) * K + k0 + vec]);
        }
    };

    const int nK = K / TKK;
    stage(0, 0);
    for (int kt = 0; kt < nK; ++kt) {
        const int cur = kt & 1;
        if (kt + 1 < nK) { stage(kt + 1, cur ^ 1); wait_async<6>(); }
        else             { wait_async<0>(); }
        __syncthreads();                           // tile kt visible to all

        v16bf af[4], bfr[4];
        #pragma unroll
        for (int i = 0; i < 4; ++i) {
            int row = waveM * 64 + i * 16 + lm;
            v8bf lo = *(const v8bf*)(&As[cur][row * LDT + g * 8]);
            v8bf hi = *(const v8bf*)(&As[cur][row * LDT + 16 + g * 8]);
            af[i] = __builtin_shufflevector(lo, hi,
                    0,1,2,3,4,5,6,7,8,9,10,11,12,13,14,15);
        }
        #pragma unroll
        for (int j = 0; j < 4; ++j) {
            int col = waveN * 64 + j * 16 + lm;
            v8bf lo = *(const v8bf*)(&Bs[cur][col * LDT + g * 8]);
            v8bf hi = *(const v8bf*)(&Bs[cur][col * LDT + 16 + g * 8]);
            bfr[j] = __builtin_shufflevector(lo, hi,
                    0,1,2,3,4,5,6,7,8,9,10,11,12,13,14,15);
        }
        #pragma unroll
        for (int i = 0; i < 4; ++i)
            #pragma unroll
            for (int j = 0; j < 4; ++j)
                acc[i][j] = __builtin_amdgcn_wmma_f32_16x16x32_bf16(
                    false, af[i], false, bfr[j], (short)0, acc[i][j],
                    false, false);

        __syncthreads();                           // done reading buf cur
    }

    // epilogue: lane owns col n = nBase+waveN*64+j*16+lm; row m = base+r+8g.
    #pragma unroll
    for (int j = 0; j < 4; ++j) {
        const int n = nBase + waveN * 64 + j * 16 + lm;
        float s, t;
        if constexpr (EPI == 0) {
            s = gamma[n] * rsqrtf(var[n] + EPSF);
            t = (bias[n] - mean[n]) * s + beta[n];
        } else {
            s = 1.f;
            t = bias[n];
        }
        #pragma unroll
        for (int i = 0; i < 4; ++i) {
            const int m0 = mBase + waveM * 64 + i * 16 + 8 * g;
            #pragma unroll
            for (int r = 0; r < 8; ++r) {
                float v = acc[i][j][r];
                size_t off = (size_t)(m0 + r) * N + n;
                if constexpr (EPI == 0) {
                    float y = fmaxf(fmaf(v, s, t), 0.f);
                    outb[off] = f2bf(y);
                } else {
                    float y = v + t;
                    outf[off] = y;
                    if (outb) outb[off] = f2bf(y);
                }
            }
        }
    }
}

// --------------------------- VQ distance argmin ----------------------------
// scores = Z[B,E] * CB^T ; dist = cnorm[n] - 2*score ; row-min via packed
// (fkey(dist)<<32 | code) and atomicMin u64.

__global__ __launch_bounds__(256)
void k_vq_gemm(const bf16_t* __restrict__ Z, const bf16_t* __restrict__ CB,
               const float* __restrict__ cnorm,
               unsigned long long* __restrict__ best)
{
    __shared__ __align__(16) bf16_t As[2][TM * LDT];
    __shared__ __align__(16) bf16_t Bs[2][TN * LDT];

    const int mBase = blockIdx.y * TM;
    const int nBase = blockIdx.x * TN;
    const int tid   = threadIdx.x;
    const int lane  = tid & 31;
    const int wave  = tid >> 5;
    const int g     = lane >> 4;
    const int lm    = lane & 15;
    const int waveM = wave >> 1;
    const int waveN = wave & 1;
    const int K     = EE;

    v8f acc[4][4];
    for (int i = 0; i < 4; ++i)
        for (int j = 0; j < 4; ++j)
            acc[i][j] = (v8f){0.f,0.f,0.f,0.f,0.f,0.f,0.f,0.f};

    auto stage = [&](int kt, int buf) {
        const int k0 = kt * TKK;
        #pragma unroll
        for (int i = 0; i < 4; ++i) {
            int lin = i * 256 + tid;
            int row = lin >> 2, vec = (lin & 3) * 8;
            cp_async16(&As[buf][row * LDT + vec],
                       &Z[(size_t)(mBase + row) * K + k0 + vec]);
        }
        #pragma unroll
        for (int i = 0; i < 2; ++i) {
            int lin = i * 256 + tid;
            int row = lin >> 2, vec = (lin & 3) * 8;
            cp_async16(&Bs[buf][row * LDT + vec],
                       &CB[(size_t)(nBase + row) * K + k0 + vec]);
        }
    };

    const int nK = K / TKK;                        // 8
    stage(0, 0);
    for (int kt = 0; kt < nK; ++kt) {
        const int cur = kt & 1;
        if (kt + 1 < nK) { stage(kt + 1, cur ^ 1); wait_async<6>(); }
        else             { wait_async<0>(); }
        __syncthreads();

        v16bf af[4], bfr[4];
        #pragma unroll
        for (int i = 0; i < 4; ++i) {
            int row = waveM * 64 + i * 16 + lm;
            v8bf lo = *(const v8bf*)(&As[cur][row * LDT + g * 8]);
            v8bf hi = *(const v8bf*)(&As[cur][row * LDT + 16 + g * 8]);
            af[i] = __builtin_shufflevector(lo, hi,
                    0,1,2,3,4,5,6,7,8,9,10,11,12,13,14,15);
        }
        #pragma unroll
        for (int j = 0; j < 4; ++j) {
            int col = waveN * 64 + j * 16 + lm;
            v8bf lo = *(const v8bf*)(&Bs[cur][col * LDT + g * 8]);
            v8bf hi = *(const v8bf*)(&Bs[cur][col * LDT + 16 + g * 8]);
            bfr[j] = __builtin_shufflevector(lo, hi,
                    0,1,2,3,4,5,6,7,8,9,10,11,12,13,14,15);
        }
        #pragma unroll
        for (int i = 0; i < 4; ++i)
            #pragma unroll
            for (int j = 0; j < 4; ++j)
                acc[i][j] = __builtin_amdgcn_wmma_f32_16x16x32_bf16(
                    false, af[i], false, bfr[j], (short)0, acc[i][j],
                    false, false);

        __syncthreads();
    }

    #pragma unroll
    for (int i = 0; i < 4; ++i) {
        #pragma unroll
        for (int r = 0; r < 8; ++r) {
            unsigned long long pk = 0xFFFFFFFFFFFFFFFFull;
            #pragma unroll
            for (int j = 0; j < 4; ++j) {
                int n = nBase + waveN * 64 + j * 16 + lm;
                float dist = cnorm[n] - 2.0f * acc[i][j][r];
                unsigned long long cand =
                    ((unsigned long long)fkey(dist) << 32) | (unsigned int)n;
                pk = (cand < pk) ? cand : pk;
            }
            for (int sh = 1; sh < 16; sh <<= 1) {   // reduce 16-lane N-group
                unsigned long long o = __shfl_xor(pk, sh, 32);
                pk = (o < pk) ? o : pk;
            }
            if (lm == 0) {
                int m = mBase + waveM * 64 + i * 16 + r + 8 * g;
                atomicMin(&best[m], pk);
            }
        }
    }
}

__global__ void k_gather(const unsigned long long* __restrict__ best,
                         const float* __restrict__ cb,
                         float* __restrict__ zq_f,
                         bf16_t* __restrict__ zq_b) {
    const int row = blockIdx.x, t = threadIdx.x;   // 256 threads = E
    unsigned int idx = (unsigned int)(best[row] & 0xFFFFFFFFu);
    float v = cb[(size_t)idx * EE + t];
    zq_f[(size_t)row * EE + t] = v;
    zq_b[(size_t)row * EE + t] = f2bf(v);
}

// ------------------------------- launcher ----------------------------------

static inline size_t alignup(size_t x) { return (x + 255) & ~(size_t)255; }

extern "C" void kernel_launch(void* const* d_in, const int* in_sizes, int n_in,
                              void* d_out, int out_size, void* d_ws, size_t ws_size,
                              hipStream_t stream) {
    (void)in_sizes; (void)n_in; (void)out_size; (void)ws_size;

    const float* x      = (const float*)d_in[0];
    const float* e_w1 = (const float*)d_in[1];   const float* e_b1 = (const float*)d_in[2];
    const float* e_g1 = (const float*)d_in[3];   const float* e_be1= (const float*)d_in[4];
    const float* e_m1 = (const float*)d_in[5];   const float* e_v1 = (const float*)d_in[6];
    const float* e_w2 = (const float*)d_in[7];   const float* e_b2 = (const float*)d_in[8];
    const float* e_g2 = (const float*)d_in[9];   const float* e_be2= (const float*)d_in[10];
    const float* e_m2 = (const float*)d_in[11];  const float* e_v2 = (const float*)d_in[12];
    const float* e_w3 = (const float*)d_in[13];  const float* e_b3 = (const float*)d_in[14];
    const float* d_w1 = (const float*)d_in[15];  const float* d_b1 = (const float*)d_in[16];
    const float* d_g1 = (const float*)d_in[17];  const float* d_be1= (const float*)d_in[18];
    const float* d_m1 = (const float*)d_in[19];  const float* d_v1 = (const float*)d_in[20];
    const float* d_w2 = (const float*)d_in[21];  const float* d_b2 = (const float*)d_in[22];
    const float* d_g2 = (const float*)d_in[23];  const float* d_be2= (const float*)d_in[24];
    const float* d_m2 = (const float*)d_in[25];  const float* d_v2 = (const float*)d_in[26];
    const float* d_w3 = (const float*)d_in[27];  const float* d_b3 = (const float*)d_in[28];
    const float* cb   = (const float*)d_in[29];

    float* out_recon = (float*)d_out;                         // [B, DIN]
    float* out_z     = out_recon + (size_t)BB * DIN;          // [B, E]
    float* out_zq    = out_z     + (size_t)BB * EE;           // [B, E]

    char* p = (char*)d_ws; size_t off = 0;
    auto take = [&](size_t bytes) { char* q = p + off; off = alignup(off + bytes); return q; };
    bf16_t* xb   = (bf16_t*)take((size_t)BB * DIN * 2);
    bf16_t* hA   = (bf16_t*)take((size_t)BB * HH  * 2);
    bf16_t* hBuf = (bf16_t*)take((size_t)BB * HH  * 2);
    bf16_t* zb   = (bf16_t*)take((size_t)BB * EE  * 2);
    bf16_t* zqb  = (bf16_t*)take((size_t)BB * EE  * 2);
    bf16_t* w1eT = (bf16_t*)take((size_t)HH  * DIN * 2);
    bf16_t* w2eT = (bf16_t*)take((size_t)HH  * HH  * 2);
    bf16_t* w3eT = (bf16_t*)take((size_t)EE  * HH  * 2);
    bf16_t* w1dT = (bf16_t*)take((size_t)HH  * EE  * 2);
    bf16_t* w2dT = (bf16_t*)take((size_t)HH  * HH  * 2);
    bf16_t* w3dT = (bf16_t*)take((size_t)DIN * HH  * 2);
    bf16_t* cbb  = (bf16_t*)take((size_t)KC  * EE  * 2);
    float*  cnrm = (float*)take((size_t)KC * 4);
    unsigned long long* best = (unsigned long long*)take((size_t)BB * 8);

    // ---- precision conversion / layout prep ----
    k_f32_to_bf16<<<(BB * DIN) / 1024, 256, 0, stream>>>(x, xb, BB * DIN);
    k_transpose_bf16<<<dim3(HH/32,  DIN/32), dim3(32,8), 0, stream>>>(e_w1, w1eT, DIN, HH);
    k_transpose_bf16<<<dim3(HH/32,  HH/32),  dim3(32,8), 0, stream>>>(e_w2, w2eT, HH,  HH);
    k_transpose_bf16<<<dim3(EE/32,  HH/32),  dim3(32,8), 0, stream>>>(e_w3, w3eT, HH,  EE);
    k_transpose_bf16<<<dim3(HH/32,  EE/32),  dim3(32,8), 0, stream>>>(d_w1, w1dT, EE,  HH);
    k_transpose_bf16<<<dim3(HH/32,  HH/32),  dim3(32,8), 0, stream>>>(d_w2, w2dT, HH,  HH);
    k_transpose_bf16<<<dim3(DIN/32, HH/32),  dim3(32,8), 0, stream>>>(d_w3, w3dT, HH,  DIN);
    k_cb_prep<<<KC, 256, 0, stream>>>(cb, cbb, cnrm);
    k_init_best<<<BB / 256, 256, 0, stream>>>(best, BB);

    const dim3 blk(256);
    // ---- encoder ----
    k_gemm<0><<<dim3(HH/TN, BB/TM), blk, 0, stream>>>(xb,  w1eT, BB, HH, DIN,
        e_b1, e_g1, e_be1, e_m1, e_v1, nullptr, hA);
    k_gemm<0><<<dim3(HH/TN, BB/TM), blk, 0, stream>>>(hA,  w2eT, BB, HH, HH,
        e_b2, e_g2, e_be2, e_m2, e_v2, nullptr, hBuf);
    k_gemm<1><<<dim3(EE/TN, BB/TM), blk, 0, stream>>>(hBuf, w3eT, BB, EE, HH,
        e_b3, nullptr, nullptr, nullptr, nullptr, out_z, zb);
    // ---- vector quantization ----
    k_vq_gemm<<<dim3(KC/TN, BB/TM), blk, 0, stream>>>(zb, cbb, cnrm, best);
    k_gather<<<BB, 256, 0, stream>>>(best, cb, out_zq, zqb);
    // ---- decoder ----
    k_gemm<0><<<dim3(HH/TN, BB/TM), blk, 0, stream>>>(zqb, w1dT, BB, HH, EE,
        d_b1, d_g1, d_be1, d_m1, d_v1, nullptr, hA);
    k_gemm<0><<<dim3(HH/TN, BB/TM), blk, 0, stream>>>(hA,  w2dT, BB, HH, HH,
        d_b2, d_g2, d_be2, d_m2, d_v2, nullptr, hBuf);
    k_gemm<1><<<dim3(DIN/TN, BB/TM), blk, 0, stream>>>(hBuf, w3dT, BB, DIN, HH,
        d_b3, nullptr, nullptr, nullptr, nullptr, out_recon, nullptr);
}
